// MultiplicativeAttention_77395310674347
// MI455X (gfx1250) — matrix operand order repaired and checked by reading, not compiled, not run on hardware
//
#include <hip/hip_runtime.h>
#include <hip/hip_bf16.h>

typedef __attribute__((ext_vector_type(16))) _Float16 v16h;
typedef __attribute__((ext_vector_type(8)))  _Float16 v8h;
typedef __attribute__((ext_vector_type(4)))  _Float16 v4h;
typedef __attribute__((ext_vector_type(8)))  float    v8f;
typedef __attribute__((ext_vector_type(4)))  float    v4f;

#define BB 8
#define CC 128
#define NN 4096
#define LDS_STRIDE (NN + 8)   /* halves; keeps v8h loads 16B-aligned, breaks bank stride */
#define WAVES 8

extern "C" __global__ __launch_bounds__(256)
void fused_attn_kernel(const float* __restrict__ q,
                       const float* __restrict__ k,
                       const float* __restrict__ v,
                       float* __restrict__ ctx,    /* [B, N, C] flat (== view b,c,h,w) */
                       float* __restrict__ attn)   /* [B, N, N] */
{
    extern __shared__ _Float16 smem[];             /* [16][LDS_STRIDE] f16 scores/attn */

    const int tid   = threadIdx.x;
    const int wave  = tid >> 5;
    const int lane  = tid & 31;
    const int lhalf = lane >> 4;                   /* 0 or 1 */
    const int lmod  = lane & 15;

    const int blk = blockIdx.x;                    /* B * (N/16) = 2048 blocks */
    const int b   = blk >> 8;
    const int i0  = (blk & 255) << 4;

    const float* qb = q + (size_t)b * CC * NN;
    const float* kb = k + (size_t)b * CC * NN;
    const float* vb = v + (size_t)b * CC * NN;

    /* ---- Preload A = Q^T tile (rows i0..i0+15, all 128 channels) into 4 frags ----
       A[m][kk] = q[c][i0+m]; lane holds M=lmod, K(l,h) = (h>>3)*16 + lhalf*8 + (h&7) */
    v16h afrag[4];
    #pragma unroll
    for (int s = 0; s < 4; ++s) {
        #pragma unroll
        for (int h = 0; h < 16; ++h) {
            int c = 32 * s + ((h >> 3) << 4) + (lhalf << 3) + (h & 7);
            afrag[s][h] = (_Float16)qb[(size_t)c * NN + i0 + lmod];
        }
    }

    /* ---- GEMM1: S[16][4096] = A x K, f32 accum, stored f16 into LDS ---- */
    for (int jt = wave; jt < NN / 16; jt += WAVES) {
        const int j0 = jt << 4;
        v8f acc = {};
        #pragma unroll
        for (int s = 0; s < 4; ++s) {
            /* B[kk][n] = key[c0+kk][j0+n]; lane: n=lmod, K = (h>>3)*16 + lhalf*8 + (h&7) */
            v16h bfrag;
            #pragma unroll
            for (int h = 0; h < 16; ++h) {
                int c = 32 * s + ((h >> 3) << 4) + (lhalf << 3) + (h & 7);
                bfrag[h] = (_Float16)kb[(size_t)c * NN + j0 + lmod];
            }
            acc = __builtin_amdgcn_wmma_f32_16x16x32_f16(
                      false, afrag[s], false, bfrag, (short)0, acc, false, false);
        }
        /* D layout: lane holds col n=lmod, rows r + 8*lhalf */
        #pragma unroll
        for (int r = 0; r < 8; ++r) {
            int m = r + (lhalf << 3);
            smem[m * LDS_STRIDE + j0 + lmod] = (_Float16)acc[r];
        }
    }
    __syncthreads();

    /* ---- Softmax over each of the 16 rows (wave w -> rows 2w, 2w+1).
       Each lane owns contiguous 4-element chunks: ds_load_b64 / b128 global stores. */
    #pragma unroll
    for (int rr = 0; rr < 2; ++rr) {
        const int row = wave * 2 + rr;
        _Float16* srow = smem + row * LDS_STRIDE;

        float mx = -3.4e38f;
        for (int j = lane * 4; j < NN; j += 128) {
            v4h c4 = *(const v4h*)(srow + j);
            mx = fmaxf(mx, fmaxf(fmaxf((float)c4[0], (float)c4[1]),
                                 fmaxf((float)c4[2], (float)c4[3])));
        }
        #pragma unroll
        for (int off = 16; off; off >>= 1)
            mx = fmaxf(mx, __shfl_xor(mx, off, 32));

        float sum = 0.f;
        for (int j = lane * 4; j < NN; j += 128) {
            v4h c4 = *(const v4h*)(srow + j);
            v4h e4;
            #pragma unroll
            for (int t = 0; t < 4; ++t) {
                float e = __expf((float)c4[t] - mx);
                e4[t] = (_Float16)e;
                sum += e;
            }
            *(v4h*)(srow + j) = e4;
        }
        #pragma unroll
        for (int off = 16; off; off >>= 1)
            sum += __shfl_xor(sum, off, 32);
        const float inv = 1.0f / sum;

        float* arow = attn + ((size_t)(b * NN + i0 + row)) * NN;
        for (int j = lane * 4; j < NN; j += 128) {
            v4h e4 = *(const v4h*)(srow + j);
            v4f p4;
            #pragma unroll
            for (int t = 0; t < 4; ++t) {
                float p = (float)e4[t] * inv;
                p4[t] = p;
                e4[t] = (_Float16)p;
            }
            *(v4h*)(srow + j) = e4;                      /* reused as GEMM2 A operand */
            __builtin_nontemporal_store(p4, (v4f*)(arow + j));  /* 537 MB stream, NT */
        }
    }
    __syncthreads();

    /* ---- GEMM2: ctx[i,c] = sum_j attn[i,j] * v[c,j]; wave w -> c-tile w ---- */
    {
        const int c0 = wave << 4;
        v8f acc = {};
        for (int j0 = 0; j0 < NN; j0 += 32) {
            /* A from LDS: K(l,h) = (h>>3)*16 + lhalf*8 + (h&7), two aligned v8h loads */
            const _Float16* arow = smem + lmod * LDS_STRIDE + j0 + (lhalf << 3);
            v8h lo = *(const v8h*)(arow);
            v8h hi = *(const v8h*)(arow + 16);
            v16h af;
            #pragma unroll
            for (int h = 0; h < 8; ++h) { af[h] = lo[h]; af[h + 8] = hi[h]; }

            /* B[kk][n] = v[c0+n][j0+kk]; lane: n=lmod, K = (h>>3)*16 + lhalf*8 + (h&7) */
            const float* vrow = vb + (size_t)(c0 + lmod) * NN + j0 + (lhalf << 3);
            v16h bf;
            #pragma unroll
            for (int h = 0; h < 16; ++h) bf[h] = (_Float16)vrow[((h >> 3) << 4) + (h & 7)];

            acc = __builtin_amdgcn_wmma_f32_16x16x32_f16(
                      false, af, false, bf, (short)0, acc, false, false);
        }
        #pragma unroll
        for (int r = 0; r < 8; ++r) {
            int m = r + (lhalf << 3);
            __builtin_nontemporal_store(
                acc[r], ctx + (size_t)(b * NN + i0 + m) * CC + c0 + lmod);
        }
    }
}

extern "C" void kernel_launch(void* const* d_in, const int* in_sizes, int n_in,
                              void* d_out, int out_size, void* d_ws, size_t ws_size,
                              hipStream_t stream) {
    (void)in_sizes; (void)n_in; (void)out_size; (void)d_ws; (void)ws_size;

    const float* q = (const float*)d_in[0];
    const float* k = (const float*)d_in[1];
    const float* v = (const float*)d_in[2];

    float* ctx  = (float*)d_out;                         /* B*C*H*W = 4,194,304 f32 */
    float* attn = (float*)d_out + (size_t)BB * CC * NN;  /* B*N*N   = 134,217,728 f32 */

    const size_t lds_bytes = (size_t)16 * LDS_STRIDE * sizeof(_Float16); /* ~128.3 KB */
    dim3 grid(BB * (NN / 16));   /* 2048 workgroups */
    dim3 block(256);             /* 8 waves (wave32) */
    fused_attn_kernel<<<grid, block, lds_bytes, stream>>>(q, k, v, ctx, attn);
}
